// GNetVerSecond_89455578841606
// MI455X (gfx1250) — matrix operand order repaired and checked by reading, not compiled, not run on hardware
//
#include <hip/hip_runtime.h>

#define BB 8
#define LL 2048
#define K1 16
#define K2 32
#define NG 32

typedef __attribute__((ext_vector_type(16))) __bf16 v16bf;
typedef __attribute__((ext_vector_type(8)))  float  v8f;

union Frag { uint4 q[2]; v16bf v; };
union Pack2 { __bf16 h[2]; unsigned int u; };

// ---------------------------------------------------------------------------
// Kernel 0: one-time weight conversion to bf16, transposed [N][K] so that a
// WMMA B-fragment for (lane n, half h) is two 16-byte runs 32 bytes apart.
//   opkT[f][k] = bf16(op_kernel[k][f])   (128 x 640)
//   WembT[n][k] = bf16(W_embed[k][n])    (32 x 128)
// ---------------------------------------------------------------------------
__global__ void convert_weights(const float* __restrict__ opk,
                                const float* __restrict__ Wemb,
                                __bf16* __restrict__ opkT,
                                __bf16* __restrict__ WembT) {
    const int stride = gridDim.x * blockDim.x;
    for (int t = blockIdx.x * blockDim.x + threadIdx.x;
         t < 128 * 640 + 32 * 128; t += stride) {
        if (t < 128 * 640) {
            const int f = t / 640, k = t % 640;
            opkT[t] = (__bf16)opk[k * 128 + f];
        } else {
            const int u = t - 128 * 640;
            const int n = u / 128, kk = u % 128;
            WembT[u] = (__bf16)Wemb[kk * 32 + n];
        }
    }
}

// ---------------------------------------------------------------------------
// Kernel 1: KNN. One wave (32 lanes) per query point; distances cached in LDS.
// 32 argmin rounds with lexicographic (dist, index) tie-break to match top_k.
// ---------------------------------------------------------------------------
__global__ void knn_kernel(const float* __restrict__ frame, int* __restrict__ idx2) {
    extern __shared__ unsigned char dynsm[];
    float* distL = (float*)dynsm;                 // 8 waves * 2048 floats = 64 KB
    const int wid  = threadIdx.x >> 5;
    const int lane = threadIdx.x & 31;
    const int q = blockIdx.x * 8 + wid;
    const int b = q >> 11;
    const int l = q & (LL - 1);
    float* dl = distL + wid * LL;
    const float* fb = frame + (size_t)b * LL * 12;
    const float cx = fb[l * 12 + 0], cy = fb[l * 12 + 1], cz = fb[l * 12 + 2];
    for (int j = lane; j < LL; j += 32) {
        float dx = fb[j * 12 + 0] - cx;
        float dy = fb[j * 12 + 1] - cy;
        float dz = fb[j * 12 + 2] - cz;
        dl[j] = dx * dx + dy * dy + dz * dz;
    }
    __syncthreads();
    int* outq = idx2 + (size_t)q * K2;
    for (int r = 0; r < K2; ++r) {
        float bv = 3.0e38f; int bi = 0x7fffffff;
        for (int t = 0; t < LL / 32; ++t) {
            int j = lane + (t << 5);
            float v = dl[j];
            if (v < bv || (v == bv && j < bi)) { bv = v; bi = j; }
        }
        #pragma unroll
        for (int m = 16; m; m >>= 1) {
            float ov = __shfl_xor(bv, m, 32);
            int   oi = __shfl_xor(bi, m, 32);
            if (ov < bv || (ov == bv && oi < bi)) { bv = ov; bi = oi; }
        }
        if (lane == 0) { outq[r] = bi; dl[bi] = 3.0e38f; }
        __syncthreads();
    }
}

// ---------------------------------------------------------------------------
// Kernel 2: stage 1. One block (256 thr = 8 waves) handles 16 points.
//  Phase A: per-(point,neighbor) thread: frame coords, Gaussians g1, attrs.
//  Phase B: T[p][g*20+h] = sum_k g1*attr -> bf16 LDS (paired b32 stores).
//  Phase C: filt = relu(T(16x640) @ op_kernel(640x128)) via WMMA bf16,
//           software-pipelined: fragment loads for chunk kc+1 are issued
//           before the wmma of chunk kc so loads overlap matrix issue.
//  Phase D: emb = relu(filt(16x128) @ W_embed(128x32)) via WMMA (waves 0-1).
//  Phase E: tiny heads beta/self/cross/node -> global workspace.
// ---------------------------------------------------------------------------
__global__ void stage1_kernel(const float* __restrict__ attrib,
                              const float* __restrict__ frame,
                              const int*   __restrict__ seq,
                              const float* __restrict__ c1,
                              const float* __restrict__ sp1,
                              const __bf16* __restrict__ opkT,
                              const float* __restrict__ opb,
                              const __bf16* __restrict__ WembT,
                              const float* __restrict__ bemb,
                              const float* __restrict__ Wbeta,  const float* __restrict__ bbeta,
                              const float* __restrict__ Wself,  const float* __restrict__ bself,
                              const float* __restrict__ Wcross, const float* __restrict__ bcross,
                              const float* __restrict__ Wnode,  const float* __restrict__ bnode,
                              const int*   __restrict__ idx2,
                              float* __restrict__ betaW, float* __restrict__ selfW,
                              float* __restrict__ crossW, float* __restrict__ nodeW) {
    extern __shared__ unsigned char dynsm[];
    // Phase A/B region (reused in phase D/E):
    float*  g1L   = (float*)dynsm;                       // 16*16*32 f32 = 32768 B
    float*  attrL = (float*)(dynsm + 32768);             // 16*16*20 f32 = 20480 B
    __bf16* Tl    = (__bf16*)(dynsm + 53248);            // 16*640 bf16  = 20480 B
    float*  sp1L  = (float*)(dynsm + 73728);             // 1568 f32     =  6272 B
    float*  c1L   = (float*)(dynsm + 80000);             // 224 f32      =   896 B
    // Reuse of region A after the WMMA main loop:
    float*  filtL = (float*)dynsm;                       // 16*128 f32   =  8192 B
    __bf16* fbL   = (__bf16*)(dynsm + 8192);             // 16*128 bf16  =  4096 B
    float*  embL  = (float*)(dynsm + 12288);             // 16*32 f32    =  2048 B

    const int tid  = threadIdx.x;
    const int w    = tid >> 5;
    const int lane = tid & 31;
    const int row  = lane & 15;   // A row / B column within tile
    const int half = lane >> 4;

    const int gp = blockIdx.x * 16;       // global point base (16 points/block)
    const int b  = gp >> 11;
    const int l0 = gp & (LL - 1);

    // stage Gaussian params
    for (int t = tid; t < 7 * 7 * NG; t += 256) sp1L[t] = sp1[t];
    for (int t = tid; t < 7 * NG;     t += 256) c1L[t]  = c1[t];
    __syncthreads();

    // ---- Phase A: thread = (p, k) with p = tid/16, k = tid%16 ----
    {
        const int p = tid >> 4;
        const int k = tid & 15;
        const int l = l0 + p;
        const int nb = idx2[((size_t)(b * LL + l)) * K2 + k];
        const float* fq = frame + (size_t)(b * LL + l) * 12;
        const float* fn = frame + (size_t)(b * LL + nb) * 12;
        const float dx = fn[0] - fq[0], dy = fn[1] - fq[1], dz = fn[2] - fq[2];
        const float dist = sqrtf(dx * dx + dy * dy + dz * dz + 1e-8f);
        const float zqx = fq[9], zqy = fq[10], zqz = fq[11];
        const float zjx = fn[9], zjy = fn[10], zjz = fn[11];
        float coords[7];
        coords[0] = dx * fq[3] + dy * fq[4] + dz * fq[5];
        coords[1] = dx * fq[6] + dy * fq[7] + dz * fq[8];
        coords[2] = dx * fq[9] + dy * fq[10] + dz * fq[11];
        coords[3] = zqx * zjx + zqy * zjy + zqz * zjz;
        coords[4] = (dx * zjx + dy * zjy + dz * zjz) / dist;
        coords[5] = (zqx * dx + zqy * dy + zqz * dz) / dist;
        const float si = (float)seq[b * LL + l];
        const float sj = (float)seq[b * LL + nb];
        coords[6] = fminf(fabsf(sj - si), 8.0f);

        for (int g = 0; g < NG; ++g) {
            float diff[7];
            #pragma unroll
            for (int d = 0; d < 7; ++d) diff[d] = coords[d] - c1L[d * NG + g];
            float s = 0.f;
            #pragma unroll
            for (int e = 0; e < 7; ++e) {
                float y = 0.f;
                #pragma unroll
                for (int d = 0; d < 7; ++d) y += diff[d] * sp1L[(d * 7 + e) * NG + g];
                s += y * y;
            }
            g1L[(p * 16 + k) * NG + g] = __expf(-0.5f * s);
        }
        const float* ap = attrib + (size_t)(b * LL + nb) * 20;
        for (int h = 0; h < 20; ++h) attrL[(p * 16 + k) * 20 + h] = ap[h];
    }
    __syncthreads();

    // ---- Phase B: T[p][g*20+h] = sum_k g1 * attr, packed bf16 b32 stores ----
    {
        unsigned int* T32 = (unsigned int*)Tl;
        for (int t = tid; t < (16 * 640) / 2; t += 256) {
            const int e0 = 2 * t;
            const int p2 = e0 / 640;
            const int gh0 = e0 % 640;
            const float* gbase = g1L + p2 * 16 * NG;
            const float* abase = attrL + p2 * 16 * 20;
            Pack2 pk;
            #pragma unroll
            for (int j = 0; j < 2; ++j) {
                const int gh = gh0 + j;
                const int g = gh / 20, h = gh % 20;
                float s = 0.f;
                #pragma unroll
                for (int k2 = 0; k2 < K1; ++k2)
                    s += gbase[k2 * NG + g] * abase[k2 * 20 + h];
                pk.h[j] = (__bf16)s;
            }
            T32[t] = pk.u;
        }
    }
    __syncthreads();

    // ---- Phase C: WMMA main loop (M=16, K=640, N=128; wave w owns N-tile w).
    //      Software pipelined: loads for kc+1 issued before wmma of kc. ----
    v8f acc = {};
    {
        const char* Tb = (const char*)Tl;
        const char* Bb = (const char*)opkT;
        const int col = w * 16 + row;
        const size_t aOff = (size_t)(row * 640 + half * 8) * 2;
        const size_t bOff = (size_t)(col * 640 + half * 8) * 2;

        Frag a0, b0, a1, b1;
        a0.q[0] = *(const uint4*)(Tb + aOff);
        a0.q[1] = *(const uint4*)(Tb + aOff + 32);
        b0.q[0] = *(const uint4*)(Bb + bOff);
        b0.q[1] = *(const uint4*)(Bb + bOff + 32);

        #pragma unroll 4
        for (int kc = 0; kc < 20; ++kc) {
            // branch-free clamped next-chunk and prefetch offsets
            const int kn = (kc + 1 < 20) ? (kc + 1) : 19;
            const int kp = (kc + 6 < 20) ? (kc + 6) : 19;
            const size_t an = aOff + (size_t)kn * 64;
            const size_t bn = bOff + (size_t)kn * 64;
            a1.q[0] = *(const uint4*)(Tb + an);
            a1.q[1] = *(const uint4*)(Tb + an + 32);
            b1.q[0] = *(const uint4*)(Bb + bn);
            b1.q[1] = *(const uint4*)(Bb + bn + 32);
            __builtin_prefetch(Bb + bOff + (size_t)kp * 64, 0, 2);
            acc = __builtin_amdgcn_wmma_f32_16x16x32_bf16(false, a0.v, false, b0.v,
                                                          (short)0, acc, false, false);
            a0 = a1; b0 = b1;
        }
    }
    __syncthreads();   // everyone done reading Tl before region A is reused

    // ---- epilogue: bias + relu -> filt (f32 + bf16 copies) ----
    {
        const int col = w * 16 + row;
        #pragma unroll
        for (int r = 0; r < 8; ++r) {
            const int m = r + (half ? 8 : 0);
            float f = fmaxf(acc[r] + opb[col], 0.f);
            filtL[m * 128 + col] = f;
            fbL[m * 128 + col]   = (__bf16)f;
        }
    }
    __syncthreads();

    // ---- Phase D: emb = relu(filt @ W_embed), waves 0-1, 4 K-chunks each ----
    if (w < 2) {
        v8f acc2 = {};
        const char* Fb = (const char*)fbL;
        const char* Wb = (const char*)WembT;
        const int col = w * 16 + row;
        #pragma unroll
        for (int kc = 0; kc < 4; ++kc) {
            Frag a, bf;
            const char* ab = Fb + (size_t)(row * 128 + kc * 32 + half * 8) * 2;
            a.q[0] = *(const uint4*)(ab);
            a.q[1] = *(const uint4*)(ab + 32);
            const char* bb = Wb + (size_t)(col * 128 + kc * 32 + half * 8) * 2;
            bf.q[0] = *(const uint4*)(bb);
            bf.q[1] = *(const uint4*)(bb + 32);
            acc2 = __builtin_amdgcn_wmma_f32_16x16x32_bf16(false, a.v, false, bf.v,
                                                           (short)0, acc2, false, false);
        }
        #pragma unroll
        for (int r = 0; r < 8; ++r) {
            const int m = r + (half ? 8 : 0);
            embL[m * 32 + col] = fmaxf(acc2[r] + bemb[col], 0.f);
        }
    }
    __syncthreads();

    // ---- Phase E: tiny heads ----
    if (tid < 16) {
        const int p = tid;
        const size_t q = (size_t)gp + p;
        float sb = bbeta[0], ss = bself[0], sc = bcross[0];
        float n0 = bnode[0], n1 = bnode[1];
        #pragma unroll
        for (int i = 0; i < 32; ++i) {
            const float e = embL[p * 32 + i];
            sb += e * Wbeta[i];
            ss += e * Wself[i];
            sc += e * Wcross[i];
            n0 += e * Wnode[i * 2 + 0];
            n1 += e * Wnode[i * 2 + 1];
        }
        betaW[q]  = fmaxf(sb, 0.f);
        selfW[q]  = fmaxf(ss, 0.f);
        crossW[q] = fmaxf(sc, 0.f);
        nodeW[q * 2 + 0] = fmaxf(n0, 0.f);
        nodeW[q * 2 + 1] = fmaxf(n1, 0.f);
    }
}

// ---------------------------------------------------------------------------
// Kernel 3: stage 2. One wave per point; lane = neighbor k (K2 == wave32).
// Softmax + aggregation entirely in shuffles.
// ---------------------------------------------------------------------------
__global__ void stage2_kernel(const float* __restrict__ frame,
                              const int*   __restrict__ seq,
                              const float* __restrict__ c2,
                              const float* __restrict__ sp2,
                              const float* __restrict__ We2,
                              const float* __restrict__ be2,
                              const int*   __restrict__ idx2,
                              const float* __restrict__ betaW,
                              const float* __restrict__ selfW,
                              const float* __restrict__ crossW,
                              const float* __restrict__ nodeW,
                              float* __restrict__ out,
                              float* __restrict__ coeffs) {
    __shared__ float sp2L[5 * 5 * NG];
    __shared__ float c2L[5 * NG];
    __shared__ float we2L[NG];
    for (int t = threadIdx.x; t < 5 * 5 * NG; t += 256) sp2L[t] = sp2[t];
    for (int t = threadIdx.x; t < 5 * NG;     t += 256) c2L[t]  = c2[t];
    for (int t = threadIdx.x; t < NG;         t += 256) we2L[t] = We2[t];
    __syncthreads();

    const int wid  = threadIdx.x >> 5;
    const int lane = threadIdx.x & 31;
    const int q = blockIdx.x * 8 + wid;
    const int b = q >> 11;
    const int nb = idx2[(size_t)q * K2 + lane];

    const float* fq = frame + (size_t)q * 12;
    const float* fn = frame + (size_t)(b * LL + nb) * 12;
    const float dx = fn[0] - fq[0], dy = fn[1] - fq[1], dz = fn[2] - fq[2];
    const float dist = sqrtf(dx * dx + dy * dy + dz * dz + 1e-8f);
    const float zqx = fq[9], zqy = fq[10], zqz = fq[11];
    const float zjx = fn[9], zjy = fn[10], zjz = fn[11];
    float coords[5];
    coords[0] = dist;
    coords[1] = zqx * zjx + zqy * zjy + zqz * zjz;
    coords[2] = (dx * zjx + dy * zjy + dz * zjz) / dist;
    coords[3] = (zqx * dx + zqy * dy + zqz * dz) / dist;
    coords[4] = fminf(fabsf((float)seq[b * LL + nb] - (float)seq[q]), 8.0f);

    float gw = be2[0];
    for (int g = 0; g < NG; ++g) {
        float diff[5];
        #pragma unroll
        for (int d = 0; d < 5; ++d) diff[d] = coords[d] - c2L[d * NG + g];
        float s = 0.f;
        #pragma unroll
        for (int e = 0; e < 5; ++e) {
            float y = 0.f;
            #pragma unroll
            for (int d = 0; d < 5; ++d) y += diff[d] * sp2L[(d * 5 + e) * NG + g];
            s += y * y;
        }
        gw += __expf(-0.5f * s) * we2L[g];
    }
    gw = fmaxf(gw, 0.f);

    const float beta_s = betaW[q] + 1e-6f;
    float logit = gw * crossW[b * LL + nb] / beta_s;
    if (lane == 0) logit += selfW[q] / beta_s;

    float mx = logit;
    #pragma unroll
    for (int m = 16; m; m >>= 1) mx = fmaxf(mx, __shfl_xor(mx, m, 32));
    const float e = __expf(logit - mx);
    float sm = e;
    #pragma unroll
    for (int m = 16; m; m >>= 1) sm += __shfl_xor(sm, m, 32);
    const float coef = e / sm;

    float o0 = coef * nodeW[(size_t)(b * LL + nb) * 2 + 0];
    float o1 = coef * nodeW[(size_t)(b * LL + nb) * 2 + 1];
    #pragma unroll
    for (int m = 16; m; m >>= 1) {
        o0 += __shfl_xor(o0, m, 32);
        o1 += __shfl_xor(o1, m, 32);
    }
    if (lane == 0) {
        out[(size_t)q * 2 + 0] = o0;
        out[(size_t)q * 2 + 1] = o1;
    }
    coeffs[(size_t)q * K2 + lane] = coef;
}

// ---------------------------------------------------------------------------
extern "C" void kernel_launch(void* const* d_in, const int* in_sizes, int n_in,
                              void* d_out, int out_size, void* d_ws, size_t ws_size,
                              hipStream_t stream) {
    (void)in_sizes; (void)n_in; (void)out_size; (void)ws_size;
    const float* attrib = (const float*)d_in[0];
    const float* frame  = (const float*)d_in[1];
    const int*   seq    = (const int*)d_in[2];
    const float* c1     = (const float*)d_in[4];
    const float* sp1    = (const float*)d_in[5];
    const float* opk    = (const float*)d_in[6];
    const float* opb    = (const float*)d_in[7];
    const float* Wemb   = (const float*)d_in[8];
    const float* bemb   = (const float*)d_in[9];
    const float* Wbeta  = (const float*)d_in[10];
    const float* bbeta  = (const float*)d_in[11];
    const float* Wself  = (const float*)d_in[12];
    const float* bself  = (const float*)d_in[13];
    const float* Wcross = (const float*)d_in[14];
    const float* bcross = (const float*)d_in[15];
    const float* Wnode  = (const float*)d_in[16];
    const float* bnode  = (const float*)d_in[17];
    const float* c2     = (const float*)d_in[18];
    const float* sp2    = (const float*)d_in[19];
    const float* We2    = (const float*)d_in[20];
    const float* be2    = (const float*)d_in[21];

    const size_t NPTS = (size_t)BB * LL;
    char* ws = (char*)d_ws;
    int*    idx2   = (int*)ws;                                   // 2,097,152 B
    float*  betaW  = (float*)(ws + NPTS * K2 * sizeof(int));
    float*  selfW  = betaW + NPTS;
    float*  crossW = selfW + NPTS;
    float*  nodeW  = crossW + NPTS;                              // +131072 B
    __bf16* opkT   = (__bf16*)(ws + 2424832);                    // 163,840 B
    __bf16* WembT  = (__bf16*)(ws + 2588672);                    //   8,192 B

    float* out    = (float*)d_out;
    float* coeffs = out + NPTS * 2;

    convert_weights<<<168, 512, 0, stream>>>(opk, Wemb, opkT, WembT);

    knn_kernel<<<NPTS / 8, 256, 8 * LL * sizeof(float), stream>>>(frame, idx2);

    stage1_kernel<<<NPTS / 16, 256, 80896, stream>>>(
        attrib, frame, seq, c1, sp1, opkT, opb, WembT, bemb,
        Wbeta, bbeta, Wself, bself, Wcross, bcross, Wnode, bnode,
        idx2, betaW, selfW, crossW, nodeW);

    stage2_kernel<<<NPTS / 8, 256, 0, stream>>>(
        frame, seq, c2, sp2, We2, be2, idx2,
        betaW, selfW, crossW, nodeW, out, coeffs);
}
